// StackedCrfDecoderABC_17600775979699
// MI455X (gfx1250) — matrix operand order repaired and checked by reading, not compiled, not run on hardware
//
#include <hip/hip_runtime.h>
#include <cstdint>

typedef _Float16 v16h __attribute__((ext_vector_type(16)));
typedef float    v8f  __attribute__((ext_vector_type(8)));

#define B_TOT   512
#define L_SEQ   1024
#define N_ST    32
#define NPACK   4
#define NCHUNK  8
#define CHUNK_LEN 128   /* L_SEQ / NCHUNK */

/* workspace layout (bytes):
   [0, 8192)                    expT column-major f16: [pack][n][k]
   [8192, 8192+512K)            row log-scales f32:    [b][chunk][i]
   [8192+512K, +8MB)            chunk operators f16:   [b][chunk][i][k]  (rows max-normalized) */
#define WS_RS_OFF   8192
#define WS_RF_OFF   (8192 + (size_t)B_TOT*NCHUNK*N_ST*4)

__device__ __forceinline__ v16h ld_frag16(const _Float16* p0, const _Float16* p1) {
    union { uint4 q[2]; v16h v; } u;
    u.q[0] = *(const uint4*)p0;
    u.q[1] = *(const uint4*)p1;
    return u.v;
}

/* stage expT[pack] as f16, column-major: cm[p][n][k] = exp(T[p][k][n]) */
__global__ void crf_prep(const float* __restrict__ T, _Float16* __restrict__ cm) {
    int idx = blockIdx.x * blockDim.x + threadIdx.x;
    if (idx >= NPACK * N_ST * N_ST) return;
    int p = idx >> 10, r = idx & 1023;
    int k = r >> 5, n = r & 31;
    cm[p * 1024 + n * 32 + k] = (_Float16)__expf(T[p * 1024 + k * 32 + n]);
}

/* Phase 1: one wave per (batch, chunk). Compose the chunk operator
   R = M_{t_lo} ... M_{t_hi} in the exp domain with per-row rescaling.
   Per step: R' = R x expT (4 x v_wmma_f32_16x16x32_f16), column-scale by
   exp(e_t), then a per-lane row renormalize (lane owns one row => no
   cross-lane ops) through LDS, emitting the next step's f16 A fragments. */
__global__ __launch_bounds__(128) void crf_phase1(
    const float*    __restrict__ emis,
    const int*      __restrict__ lengths,
    const _Float16* __restrict__ expT_cm,
    _Float16*       __restrict__ Rf,
    float*          __restrict__ rsOut)
{
    __shared__ float sD[4][N_ST * 36];                       /* f32 scratch, stride 36 floats (16B aligned, conflict-free) */
    __shared__ __align__(16) _Float16 sA[4][N_ST * 40];      /* normalized rows f16, stride 40 halves = 80B */

    const int lane = threadIdx.x & 31;
    const int wb   = threadIdx.x >> 5;
    const int wid  = blockIdx.x * 4 + wb;                    /* 0..4095 */
    const int b    = wid >> 3;
    const int c    = wid & (NCHUNK - 1);
    const int pack = b & (NPACK - 1);
    const int lh   = lane >> 4;
    const int lm   = lane & 15;

    float*     Df = sD[wb];
    _Float16*  Ah = sA[wb];

    /* B operand (expT) fragments: col n = 16*nt + lm, element e -> K = e + 16*lh */
    const _Float16* cm = expT_cm + pack * (N_ST * N_ST);
    const v16h Bf0 = ld_frag16(cm + lm * 32 + 16 * lh,        cm + lm * 32 + 16 * lh + 8);
    const v16h Bf1 = ld_frag16(cm + (16 + lm) * 32 + 16 * lh, cm + (16 + lm) * 32 + 16 * lh + 8);

    /* identity init: lane owns row `lane` */
    {
        uint4 z; z.x = z.y = z.z = z.w = 0u;
        uint4* wp = (uint4*)(Ah + lane * 40);
        wp[0] = z; wp[1] = z; wp[2] = z; wp[3] = z;
        Ah[lane * 40 + lane] = (_Float16)1.0f;
    }
    float rs = 0.0f;

    const int len  = lengths[b];
    const int t_lo = c * CHUNK_LEN + 1;
    int t_hi = (c + 1) * CHUNK_LEN;
    if (t_hi > len - 1) t_hi = len - 1;                      /* inactive steps == identity */

    const v8f cz = {};
    const float* eb  = emis + (size_t)b * L_SEQ * N_ST;
    const float* ept = eb + (size_t)t_lo * N_ST;             /* running emission row pointer */

    for (int t = t_lo; t <= t_hi; ++t, ept += N_ST) {
        /* A fragments: row = 16*mt + lm, element e -> K = e + 16*lh */
        v16h A0 = ld_frag16(Ah + lm * 40 + 16 * lh,        Ah + lm * 40 + 16 * lh + 8);
        v16h A1 = ld_frag16(Ah + (16 + lm) * 40 + 16 * lh, Ah + (16 + lm) * 40 + 16 * lh + 8);

        v8f d00 = __builtin_amdgcn_wmma_f32_16x16x32_f16(false, A0, false, Bf0, (short)0, cz, false, false);
        v8f d01 = __builtin_amdgcn_wmma_f32_16x16x32_f16(false, A0, false, Bf1, (short)0, cz, false, false);
        v8f d10 = __builtin_amdgcn_wmma_f32_16x16x32_f16(false, A1, false, Bf0, (short)0, cz, false, false);
        v8f d11 = __builtin_amdgcn_wmma_f32_16x16x32_f16(false, A1, false, Bf1, (short)0, cz, false, false);

        float e0 = __expf(ept[lm]);
        float e1 = __expf(ept[16 + lm]);

        /* branchless prefetch ~4 timesteps (512B) ahead, clamped to the
           last row this wave touches: scalar min, no EXEC manipulation. */
        int tp = t + 4; tp = (tp > t_hi) ? t_hi : tp;
        __builtin_prefetch(eb + (size_t)tp * N_ST + lane, 0, 3);

        /* D layout: vgpr r -> row 16*mt + r + 8*lh, col 16*nt + lm; fold column scale here */
#pragma unroll
        for (int r = 0; r < 8; ++r) {
            int row = r + 8 * lh;
            Df[row * 36 + lm]             = d00[r] * e0;
            Df[row * 36 + 16 + lm]        = d01[r] * e1;
            Df[(16 + row) * 36 + lm]      = d10[r] * e0;
            Df[(16 + row) * 36 + 16 + lm] = d11[r] * e1;
        }

        /* per-lane row rescale: lane owns row `lane` (no cross-lane ops) */
        float rowv[32];
        const float4* rp = (const float4*)(Df + lane * 36);
#pragma unroll
        for (int q = 0; q < 8; ++q) {
            float4 f = rp[q];
            rowv[4 * q + 0] = f.x; rowv[4 * q + 1] = f.y;
            rowv[4 * q + 2] = f.z; rowv[4 * q + 3] = f.w;
        }
        float mx = rowv[0];
#pragma unroll
        for (int j = 1; j < 32; ++j) mx = fmaxf(mx, rowv[j]);
        rs += __logf(mx);
        float inv = 1.0f / mx;
        union { _Float16 h[32]; uint4 q[4]; } hr;
#pragma unroll
        for (int j = 0; j < 32; ++j) hr.h[j] = (_Float16)(rowv[j] * inv);
        uint4* wp = (uint4*)(Ah + lane * 40);
#pragma unroll
        for (int q = 0; q < 4; ++q) wp[q] = hr.q[q];
    }

    /* emit chunk operator (rows normalized f16) + row log-scales */
    const size_t o = (size_t)(b * NCHUNK + c);
    uint4* gq = (uint4*)(Rf + o * (N_ST * N_ST) + lane * N_ST);
    const uint4* lq = (const uint4*)(Ah + lane * 40);
#pragma unroll
    for (int q = 0; q < 4; ++q) gq[q] = lq[q];
    rsOut[o * N_ST + lane] = rs;
}

__device__ __forceinline__ float wave_max(float x) {
#pragma unroll
    for (int o = 16; o > 0; o >>= 1) x = fmaxf(x, __shfl_xor(x, o, 32));
    return x;
}
__device__ __forceinline__ float wave_sum(float x) {
#pragma unroll
    for (int o = 16; o > 0; o >>= 1) x += __shfl_xor(x, o, 32);
    return x;
}

/* Phase 2: one wave per batch. Fold the 8 chunk operators into alpha,
   fuse the tag-path score gather, emit scores - logZ. lane = state index. */
__global__ __launch_bounds__(128) void crf_phase2(
    const float*    __restrict__ emis,
    const float*    __restrict__ T,
    const float*    __restrict__ startT,
    const float*    __restrict__ endT,
    const int*      __restrict__ tags,
    const int*      __restrict__ lengths,
    const _Float16* __restrict__ Rf,
    const float*    __restrict__ rs,
    float*          __restrict__ out)
{
    const int wave = (blockIdx.x * blockDim.x + threadIdx.x) >> 5;  /* 0..511 */
    const int lane = threadIdx.x & 31;
    const int b    = wave;
    const int pack = b & (NPACK - 1);
    const int len  = lengths[b];
    const int* tg  = tags + b * L_SEQ;

    /* ---- path score ---- */
    float sc = 0.0f;
    for (int t = lane; t < L_SEQ; t += 32) {
        int tt = tg[t];
        if (t < len)     sc += emis[((size_t)b * L_SEQ + t) * N_ST + tt];
        if (t + 1 < len) sc += T[pack * 1024 + tt * 32 + tg[t + 1]];
    }
    sc = wave_sum(sc);
    sc += startT[pack * 32 + tg[0]] + endT[pack * 32 + tg[len - 1]];

    /* ---- forward: alpha (lane = state) through chunk operators ---- */
    float alpha = startT[pack * 32 + lane] + emis[(size_t)b * L_SEQ * N_ST + lane];
#pragma unroll 1
    for (int c = 0; c < NCHUNK; ++c) {
        const size_t o = (size_t)(b * NCHUNK + c);
        float x = alpha + rs[o * N_ST + lane];
        float m = wave_max(x);
        float w = __expf(x - m);
        const _Float16* Rc = Rf + o * (N_ST * N_ST);
        float v = 0.0f;
#pragma unroll
        for (int i = 0; i < 32; ++i) {
            float wi = __shfl(w, i, 32);
            v += wi * (float)Rc[i * 32 + lane];
        }
        alpha = __logf(v) + m;
    }
    alpha += endT[pack * 32 + lane];
    float m2 = wave_max(alpha);
    float s  = wave_sum(__expf(alpha - m2));
    float logZ = __logf(s) + m2;

    if (lane == 0) out[b] = sc - logZ;
}

extern "C" void kernel_launch(void* const* d_in, const int* in_sizes, int n_in,
                              void* d_out, int out_size, void* d_ws, size_t ws_size,
                              hipStream_t stream) {
    const float* emissions   = (const float*)d_in[0];
    const float* transitions = (const float*)d_in[1];
    const float* startT      = (const float*)d_in[2];
    const float* endT        = (const float*)d_in[3];
    const int*   tags        = (const int*)d_in[4];
    const int*   lengths     = (const int*)d_in[5];
    float*       out         = (float*)d_out;

    char* ws = (char*)d_ws;
    _Float16* expT_cm = (_Float16*)(ws);
    float*    rsBuf   = (float*)(ws + WS_RS_OFF);
    _Float16* Rf      = (_Float16*)(ws + WS_RF_OFF);

    crf_prep<<<16, 256, 0, stream>>>(transitions, expT_cm);
    crf_phase1<<<(B_TOT * NCHUNK) / 4, 128, 0, stream>>>(emissions, lengths, expT_cm, Rf, rsBuf);
    crf_phase2<<<B_TOT / 4, 128, 0, stream>>>(emissions, transitions, startT, endT,
                                              tags, lengths, Rf, rsBuf, out);
}
